// HydModelNet_76922864271845
// MI455X (gfx1250) — matrix operand order repaired and checked by reading, not compiled, not run on hardware
//
#include <hip/hip_runtime.h>
#include <hip/hip_bf16.h>

typedef __attribute__((ext_vector_type(16))) _Float16 v16h;
typedef __attribute__((ext_vector_type(8)))  float    v8f;

#define STEPS     2048
#define BATCH     512
#define INPUT_DIM 16
#define ROWSTRIDE 17     // 1 + INPUT_DIM
#define HIDDEN    128
#define HEADN     96     // cols 0..8 = W_in rows, 16..95 = W_out rows, rest zero

// d_ws layout (bytes). Fragments are contiguous: B0 | B1 | Bh.
#define WS_B0   0        //  4096 halves :  8 frags (tile)       of layer-0 B
#define WS_B1   8192     // 16384 halves : 32 frags (tile,kb)    of layer-1 B
#define WS_BH   40960    // 12288 halves : 24 frags (tile,kb)    of head B
#define WS_BASE 65536    //   512 floats : baseflow per batch col

// LDS fragment offsets inside sW (in halves)
#define LB0 0
#define LB1 4096
#define LBH 20480

// ---------------------------------------------------------------------------
// Kernel 1: swizzle f32 weights into CDNA5 WMMA B-fragment layout (f16).
// B element (k, n):  lane = (n & 15) + 16*((k >> 3) & 1), e = (k & 7) + 8*((k & 31) >= 16).
// Fragment = 32 lanes x 16 halves (contiguous per lane -> b128 loads later).
// ---------------------------------------------------------------------------
__global__ void frag_init(const float* __restrict__ W0, const float* __restrict__ W1,
                          const float* __restrict__ W_in, const float* __restrict__ W_out,
                          _Float16* __restrict__ ws) {
  int i = blockIdx.x * blockDim.x + threadIdx.x;
  const int nB0 = 8 * 512, nB1 = 32 * 512, nBh = 24 * 512;
  if (i < nB0) {                                   // layer 0: K=24 padded to 32
    int tile = i / 512, r = i % 512, lane = r / 16, e = r % 16;
    int n = tile * 16 + (lane & 15);
    int k = ((lane >= 16) ? 8 : 0) + (e & 7) + ((e >= 8) ? 16 : 0);
    float v = (k < 24) ? W0[n * 24 + k] : 0.f;
    ws[WS_B0 / 2 + i] = (_Float16)v;
  } else if (i < nB0 + nB1) {                      // layer 1: 128x128
    int j = i - nB0;
    int frag = j / 512, r = j % 512, lane = r / 16, e = r % 16;
    int tile = frag / 4, kb = frag % 4;
    int n = tile * 16 + (lane & 15);
    int k = kb * 32 + ((lane >= 16) ? 8 : 0) + (e & 7) + ((e >= 8) ? 16 : 0);
    ws[WS_B1 / 2 + j] = (_Float16)W1[n * 128 + k];
  } else if (i < nB0 + nB1 + nBh) {                // heads: 9 (W_in) + 80 (W_out), padded N=96
    int j = i - nB0 - nB1;
    int frag = j / 512, r = j % 512, lane = r / 16, e = r % 16;
    int tile = frag / 4, kb = frag % 4;
    int nh = tile * 16 + (lane & 15);
    int k = kb * 32 + ((lane >= 16) ? 8 : 0) + (e & 7) + ((e >= 8) ? 16 : 0);
    float v = (nh < 9) ? W_in[nh * 128 + k]
            : (nh < 16) ? 0.f : W_out[(nh - 16) * 128 + k];
    ws[WS_BH / 2 + j] = (_Float16)v;
  }
}

// ---------------------------------------------------------------------------
// Kernel 2: 25th percentile of flow over time per batch column (bitonic sort).
// ---------------------------------------------------------------------------
__global__ void baseflow_kernel(const float* __restrict__ hyd, float* __restrict__ baseflow) {
  __shared__ float sv[STEPS];
  int b = blockIdx.x;
  for (int i = threadIdx.x; i < STEPS; i += blockDim.x)
    sv[i] = hyd[(size_t)i * BATCH * ROWSTRIDE + (size_t)b * ROWSTRIDE];
  __syncthreads();
  for (int k = 2; k <= STEPS; k <<= 1) {
    for (int j = k >> 1; j > 0; j >>= 1) {
      for (int i = threadIdx.x; i < STEPS; i += blockDim.x) {
        int ixj = i ^ j;
        if (ixj > i) {
          bool up = ((i & k) == 0);
          float a = sv[i], c = sv[ixj];
          if ((a > c) == up) { sv[i] = c; sv[ixj] = a; }
        }
      }
      __syncthreads();
    }
  }
  if (threadIdx.x == 0)
    baseflow[b] = 0.25f * sv[511] + 0.75f * sv[512];   // pos = 0.25*(2047) = 511.75
}

// ---------------------------------------------------------------------------
// Kernel 3: persistent fused scan. 32 WGs x 16 batch rows, 128 thr = 4 waves.
// All weight fragments LDS-resident; x_t software-pipelined one step ahead.
// ---------------------------------------------------------------------------
__device__ __forceinline__ float sigmoidf_(float x) { return 1.f / (1.f + __expf(-x)); }

__global__ __launch_bounds__(128)
void hyd_scan(const float* __restrict__ hyd,
              const float* __restrict__ b0g, const float* __restrict__ b1g,
              const float* __restrict__ bing, const float* __restrict__ boutg,
              const _Float16* __restrict__ wsf, const float* __restrict__ baseflow,
              float* __restrict__ out) {
  __shared__ alignas(32) _Float16 sW[64 * 512];    // 64 KB: B0(8) | B1(32) | Bh(24) frags
  __shared__ alignas(32) _Float16 sA1[4 * 512];    //  4 KB layer-1 input A fragments
  __shared__ alignas(32) _Float16 sA2[4 * 512];    //  4 KB heads input A fragments
  __shared__ float sHead[16 * HEADN];              //  6 KB head pre-activations
  __shared__ float sStores[16 * 8];
  __shared__ float sX[2][16][INPUT_DIM];           //  2 KB double-buffered x_t stage
  __shared__ float sBias0[HIDDEN], sBias1[HIDDEN], sBiasH[HEADN];

  const int tid  = threadIdx.x;
  const int wave = tid >> 5;
  const int lane = tid & 31;
  const int mA   = lane & 15;          // batch row this lane serves in A layout
  const int hiL  = lane >> 4;          // 0: lanes 0-15, 1: lanes 16-31
  const int brow = blockIdx.x * 16;    // batch base of this WG

  // x-prefetch mapping: thread loads 2 of the 256 (row, feature) elements
  const int pr  = (tid * 2) >> 4;      // row 0..15
  const int pf  = (tid * 2) & 15;      // feature 0..14 (even)

  // ---- one-time: stage weight fragments + biases + x(t=0) into LDS ----
  {
    const uint4* src = (const uint4*)wsf;          // B0|B1|Bh contiguous = 4096 uint4
    uint4* dst = (uint4*)sW;
    for (int i = tid; i < 4096; i += 128) dst[i] = src[i];
    sBias0[tid] = b0g[tid];
    sBias1[tid] = b1g[tid];
    if (tid < HEADN)
      sBiasH[tid] = (tid < 9) ? bing[tid] : (tid < 16 ? 0.f : boutg[tid - 16]);
    const float* r0 = hyd + (size_t)(brow + pr) * ROWSTRIDE + 1;
    sX[0][pr][pf]     = r0[pf];
    sX[0][pr][pf + 1] = r0[pf + 1];
  }
  // ---- stores state: thread (m,j) owns stores[m][j]; init col 1 = 1 ----
  const int em = tid >> 3, ej = tid & 7;
  float st = (ej == 1) ? 1.f : 0.f;
  sStores[em * 8 + ej] = st;
  __syncthreads();

  const int t0 = wave * 2, t1 = wave * 2 + 1;      // this wave's N-tiles

  for (int t = 0; t < STEPS; ++t) {
    const int cur = t & 1, nxt = cur ^ 1;
    // ---------- build A0 fragment: 16x32 = [x_t(16) | 0.01*stores(8) | 0(8)] ----------
    v16h a0;
    if (!hiL) {
      #pragma unroll
      for (int e = 0; e < 8; ++e) a0[e]     = (_Float16)sX[cur][mA][e];      // K = 0..7
      #pragma unroll
      for (int e = 0; e < 8; ++e) a0[8 + e] = (_Float16)(0.01f * sStores[mA * 8 + e]); // K=16..23
    } else {
      #pragma unroll
      for (int e = 0; e < 8; ++e) a0[e]     = (_Float16)sX[cur][mA][8 + e];  // K = 8..15
      #pragma unroll
      for (int e = 0; e < 8; ++e) a0[8 + e] = (_Float16)0.f;                 // K = 24..31 pad
    }
    // ---------- software pipeline: issue x(t+1) loads now, commit at step end ----------
    const int tt = (t + 1 < STEPS) ? t + 1 : t;
    const float* rn = hyd + (size_t)tt * BATCH * ROWSTRIDE + (size_t)(brow + pr) * ROWSTRIDE + 1;
    float xpf0 = rn[pf];
    float xpf1 = rn[pf + 1];
    // ---------- layer 0: h1 = relu(A0 x B0 + b0) ----------
    v16h bf0 = *(const v16h*)&sW[LB0 + t0 * 512 + lane * 16];
    v16h bf1 = *(const v16h*)&sW[LB0 + t1 * 512 + lane * 16];
    v8f c0 = {}; v8f c1 = {};
    c0 = __builtin_amdgcn_wmma_f32_16x16x32_f16(false, a0, false, bf0, (short)0, c0, false, false);
    c1 = __builtin_amdgcn_wmma_f32_16x16x32_f16(false, a0, false, bf1, (short)0, c1, false, false);
    #pragma unroll
    for (int half = 0; half < 2; ++half) {
      const v8f& c = half ? c1 : c0;
      int n = (half ? t1 : t0) * 16 + mA;          // hidden column -> next layer K
      int kb = n >> 5, kp = n & 31;
      int lh = (kp >> 3) & 1;
      int e  = (kp & 7) + ((kp >= 16) ? 8 : 0);
      #pragma unroll
      for (int vi = 0; vi < 8; ++vi) {
        int mm = vi + hiL * 8;
        float v = c[vi] + sBias0[n];
        v = v > 0.f ? v : 0.f;
        sA1[kb * 512 + (mm + lh * 16) * 16 + e] = (_Float16)v;  // A-fragment scatter
      }
    }
    __syncthreads();                               // (1) h1 visible
    // ---------- layer 1: h2 = relu(A1 x B1 + b1) ----------
    v8f d0 = {}; v8f d1 = {};
    #pragma unroll
    for (int kb = 0; kb < 4; ++kb) {
      v16h af  = *(const v16h*)&sA1[kb * 512 + lane * 16];
      v16h bb0 = *(const v16h*)&sW[LB1 + (t0 * 4 + kb) * 512 + lane * 16];
      v16h bb1 = *(const v16h*)&sW[LB1 + (t1 * 4 + kb) * 512 + lane * 16];
      d0 = __builtin_amdgcn_wmma_f32_16x16x32_f16(false, af, false, bb0, (short)0, d0, false, false);
      d1 = __builtin_amdgcn_wmma_f32_16x16x32_f16(false, af, false, bb1, (short)0, d1, false, false);
    }
    #pragma unroll
    for (int half = 0; half < 2; ++half) {         // scatter straight to sA2 (no drain barrier)
      const v8f& c = half ? d1 : d0;
      int n = (half ? t1 : t0) * 16 + mA;
      int kb = n >> 5, kp = n & 31;
      int lh = (kp >> 3) & 1;
      int e  = (kp & 7) + ((kp >= 16) ? 8 : 0);
      #pragma unroll
      for (int vi = 0; vi < 8; ++vi) {
        int mm = vi + hiL * 8;
        float v = c[vi] + sBias1[n];
        v = v > 0.f ? v : 0.f;
        sA2[kb * 512 + (mm + lh * 16) * 16 + e] = (_Float16)v;
      }
    }
    __syncthreads();                               // (2) h2 visible
    // ---------- heads: logits (tile 0) + gate pre-acts (tiles 1..5), N=96 ----------
    #pragma unroll
    for (int pass = 0; pass < 2; ++pass) {
      int ht = (pass == 0) ? wave : (wave < 2 ? 4 + wave : -1);
      if (ht >= 0) {                               // wave-uniform; EXEC stays all-1s
        v8f hc = {};
        #pragma unroll
        for (int kb = 0; kb < 4; ++kb) {
          v16h af = *(const v16h*)&sA2[kb * 512 + lane * 16];
          v16h bh = *(const v16h*)&sW[LBH + (ht * 4 + kb) * 512 + lane * 16];
          hc = __builtin_amdgcn_wmma_f32_16x16x32_f16(false, af, false, bh, (short)0, hc, false, false);
        }
        int n = ht * 16 + mA;
        #pragma unroll
        for (int vi = 0; vi < 8; ++vi)
          sHead[(vi + hiL * 8) * HEADN + n] = hc[vi] + sBiasH[n];
      }
    }
    __syncthreads();                               // (3) head pre-acts visible
    // ---------- elementwise store routing: 8 threads per batch row ----------
    {
      int gb = brow + em;
      const float* hrow = &sHead[em * HEADN];
      float rain = sX[cur][em][0];                 // feature 0 (RAIN_IDX)
      // softmax over 9 logits, add a[1+j]*rain to this thread's store
      float mx = hrow[0];
      #pragma unroll
      for (int i = 1; i < 9; ++i) mx = fmaxf(mx, hrow[i]);
      float den = 0.f;
      #pragma unroll
      for (int i = 0; i < 9; ++i) den += __expf(hrow[i] - mx);
      st += (__expf(hrow[ej + 1] - mx) / den) * rain;
      // sequential routing loop: fb = b[:,d*8+j]*st; st -= fb; st[d] += sum(fb)
      #pragma unroll
      for (int d = 0; d < 8; ++d) {
        float fb = sigmoidf_(hrow[16 + d * 8 + ej]) * st;
        float tot = fb;
        tot += __shfl_xor(tot, 1);
        tot += __shfl_xor(tot, 2);
        tot += __shfl_xor(tot, 4);                 // sum within aligned 8-group
        st -= fb;
        if (ej == d) st += tot;
      }
      st *= (1.f - sigmoidf_(hrow[16 + 64 + ej])); // escape
      float bfl = sigmoidf_(hrow[16 + 72 + ej]);   // flow gate
      float fd = bfl * st;
      st -= fd;
      float ft = fd;
      ft += __shfl_xor(ft, 1);
      ft += __shfl_xor(ft, 2);
      ft += __shfl_xor(ft, 4);
      if (ej == 0) out[(size_t)t * BATCH + gb] = ft;
      if (t == 0 && ej == 2)                       // is_first baseflow correction (SLOW=2)
        st = baseflow[gb] / fmaxf(bfl, 1e-5f);
      sStores[em * 8 + ej] = st;
    }
    // commit pipelined x(t+1) (loads issued ~a full step ago)
    sX[nxt][pr][pf]     = xpf0;
    sX[nxt][pr][pf + 1] = xpf1;
    __syncthreads();                               // (4) stores + x(t+1) visible
  }
}

// ---------------------------------------------------------------------------
extern "C" void kernel_launch(void* const* d_in, const int* in_sizes, int n_in,
                              void* d_out, int out_size, void* d_ws, size_t ws_size,
                              hipStream_t stream) {
  const float* hyd   = (const float*)d_in[0];
  const float* W0    = (const float*)d_in[1];
  const float* b0    = (const float*)d_in[2];
  const float* W1    = (const float*)d_in[3];
  const float* b1    = (const float*)d_in[4];
  const float* W_in  = (const float*)d_in[5];
  const float* b_in  = (const float*)d_in[6];
  const float* W_out = (const float*)d_in[7];
  const float* b_out = (const float*)d_in[8];
  float*       outp  = (float*)d_out;
  _Float16*    wsh   = (_Float16*)d_ws;
  float*       bflow = (float*)((char*)d_ws + WS_BASE);

  hipLaunchKernelGGL(frag_init, dim3(128), dim3(256), 0, stream, W0, W1, W_in, W_out, wsh);
  hipLaunchKernelGGL(baseflow_kernel, dim3(BATCH), dim3(256), 0, stream, hyd, bflow);
  hipLaunchKernelGGL(hyd_scan, dim3(BATCH / 16), dim3(128), 0, stream,
                     hyd, b0, b1, b_in, b_out, wsh, bflow, outp);
}
